// LinearSelfAttention_12756052869472
// MI455X (gfx1250) — matrix operand order repaired and checked by reading, not compiled, not run on hardware
//
#include <hip/hip_runtime.h>
#include <hip/hip_bf16.h>

// Problem constants (from reference)
#define B_    4
#define S_    4096
#define D_    1024
#define H_    16
#define HD_   64
#define MTOT  (B_ * S_)      // 16384 rows
#define KV_SPLIT 8

typedef unsigned short u16;
typedef __attribute__((ext_vector_type(16))) __bf16 v16bf;
typedef __attribute__((ext_vector_type(8)))  float  v8f;
typedef __attribute__((ext_vector_type(4)))  int    v4i_;

// CDNA5 async global->LDS path (ASYNCcnt), guarded so compile can't break.
#if defined(__HIP_DEVICE_COMPILE__) && \
    __has_builtin(__builtin_amdgcn_global_load_async_to_lds_b128) && \
    __has_builtin(__builtin_amdgcn_s_wait_asynccnt)
#define USE_ASYNC 1
#else
#define USE_ASYNC 0
#endif

typedef __attribute__((address_space(1))) v4i_ as1_v4i;   // global int4
typedef __attribute__((address_space(3))) v4i_ as3_v4i;   // LDS int4

__device__ __forceinline__ void copy16_async(void* lds, const void* gmem) {
#if USE_ASYNC
    __builtin_amdgcn_global_load_async_to_lds_b128(
        (as1_v4i*)gmem, (as3_v4i*)lds, 0, 0);
#else
    *(uint4*)lds = *(const uint4*)gmem;          // sync fallback (DScnt path)
#endif
}
template <int N>
__device__ __forceinline__ void wait_async() {
#if USE_ASYNC
    __builtin_amdgcn_s_wait_asynccnt(N);
#endif
}

union Frag { v16bf v; uint4 q[2]; };

// ---- scalar conversions -------------------------------------------------
__device__ __forceinline__ u16 f2bf(float x) {
    union { float f; unsigned u; } v; v.f = x;
    unsigned r = v.u + 0x7FFFu + ((v.u >> 16) & 1u);   // round-to-nearest-even
    return (u16)(r >> 16);
}
__device__ __forceinline__ float bf2f(u16 x) {
    union { unsigned u; float f; } v; v.u = ((unsigned)x) << 16;
    return v.f;
}

// ---- WMMA fragment loads (wave32, 16x16x32 bf16) ------------------------
// A: 16x32 (MxK) row-major. lane half 0 holds K {0..7,16..23}, half 1 {8..15,24..31}.
__device__ __forceinline__ v16bf load_a(const u16* __restrict__ p, int ld) {
    int lane = threadIdx.x & 31;
    const u16* row = p + (lane & 15) * ld + (lane >> 4) * 8;
    Frag f;
    f.q[0] = *(const uint4*)(row);
    f.q[1] = *(const uint4*)(row + 16);
    return f.v;
}
// B: 32x16 (KxN); source stored as N rows x K contiguous. lane = N,
// lane half 0 holds K {0..7,16..23}, half 1 {8..15,24..31}.
__device__ __forceinline__ v16bf load_b(const u16* __restrict__ p, int ld) {
    int lane = threadIdx.x & 31;
    const u16* row = p + (lane & 15) * ld + (lane >> 4) * 8;
    Frag f;
    f.q[0] = *(const uint4*)(row);
    f.q[1] = *(const uint4*)(row + 16);
    return f.v;
}
__device__ __forceinline__ v8f wmma_bf16(v16bf a, v16bf b, v8f c) {
    return __builtin_amdgcn_wmma_f32_16x16x32_bf16(false, a, false, b,
                                                   (short)0, c, false, false);
}

// ---- fp32 -> bf16 bulk convert (vectorized, HBM stream) ------------------
__global__ void cvt_kernel(const float* __restrict__ src, u16* __restrict__ dst, int n) {
    int i = (blockIdx.x * blockDim.x + threadIdx.x) * 8;
    int stride = gridDim.x * blockDim.x * 8;
    for (; i < n; i += stride) {
        float4 f0 = *(const float4*)(src + i);
        float4 f1 = *(const float4*)(src + i + 4);
        union { u16 h[8]; uint4 q; } u;
        u.h[0] = f2bf(f0.x); u.h[1] = f2bf(f0.y); u.h[2] = f2bf(f0.z); u.h[3] = f2bf(f0.w);
        u.h[4] = f2bf(f1.x); u.h[5] = f2bf(f1.y); u.h[6] = f2bf(f1.z); u.h[7] = f2bf(f1.w);
        *(uint4*)(dst + i) = u.q;
    }
}

__global__ void zero_kernel(float* __restrict__ p, int n) {
    int i = blockIdx.x * blockDim.x + threadIdx.x;
    if (i < n) p[i] = 0.f;
}

// ---- projection GEMM: C[m,n] = sum_k A[m,k]*W[n,k] + bias[n] ------------
// Block: 256 thr = 8 waves; block tile 256(M) x 64(N); wave tile 32x64.
// Shared W tile (64x32 bf16 = 4KB) double-buffered in LDS via async loads;
// A fragments pipelined in registers straight from global (L2-resident).
// mode 0: q  -> elu+1, layout [B,H,S,HD]   bf16
// mode 1: k  -> elu+1, layout [B,H,HD,S]   bf16 (transposed)
// mode 2: v  ->        layout [B,H,HD,S]   bf16 (transposed)
// mode 3: y  ->        layout [B,S,D]      f32  (final output)
__global__ __launch_bounds__(256) void proj_kernel(
    const u16* __restrict__ A, const u16* __restrict__ W,
    const float* __restrict__ bias, void* __restrict__ out, int mode)
{
    __shared__ __align__(16) u16 Bs[2][64 * 32];   // 8KB double buffer

    const int tid  = threadIdx.x;
    const int wave = tid >> 5;
    const int lane = tid & 31;
    const int m0 = blockIdx.y * 256 + wave * 32;
    const int n0 = blockIdx.x * 64;

    // B-stage: each of 256 threads copies 16B per k-step (64 rows x 32 cols)
    const int brow = tid >> 2;            // 0..63
    const int bcol = (tid & 3) * 8;       // 0,8,16,24
    const u16* bsrc = W + (size_t)(n0 + brow) * D_ + bcol;
    const int boff = brow * 32 + bcol;

    const u16* arow = A + (size_t)m0 * D_;

    // prologue: stage k-block 0, preload A frags for k-block 0
    copy16_async(&Bs[0][boff], bsrc);
    v16bf a0 = load_a(arow, D_);
    v16bf a1 = load_a(arow + 16 * D_, D_);

    v8f acc[2][4] = {};
    int cur = 0;
    for (int i = 0; i < D_ / 32; ++i) {
        const int kk_next = (i + 1) * 32;
        if (kk_next < D_) {
            copy16_async(&Bs[cur ^ 1][boff], bsrc + kk_next);
            wait_async<1>();              // previous stage complete (in-order)
        } else {
            wait_async<0>();
        }
        __syncthreads();                  // B tile visible to all waves

        // pipeline next A frags under this step's WMMAs
        v16bf an0 = a0, an1 = a1;
        if (kk_next < D_) {
            an0 = load_a(arow + kk_next, D_);
            an1 = load_a(arow + 16 * D_ + kk_next, D_);
        }

        const u16* bb = &Bs[cur][0];
        #pragma unroll
        for (int t = 0; t < 4; ++t) {
            v16bf b = load_b(bb + t * 16 * 32, 32);
            acc[0][t] = wmma_bf16(a0, b, acc[0][t]);
            acc[1][t] = wmma_bf16(a1, b, acc[1][t]);
        }
        __syncthreads();                  // everyone done reading Bs[cur]
        a0 = an0; a1 = an1;
        cur ^= 1;
    }

    // C layout: lane -> column n = lane&15 ; row m = (lane>>4)*8 + i
    const int nl = lane & 15;
    const int mr = (lane >> 4) * 8;
    #pragma unroll
    for (int rr = 0; rr < 2; ++rr) {
        #pragma unroll
        for (int t = 0; t < 4; ++t) {
            int n = n0 + t * 16 + nl;
            float bn = bias[n];
            #pragma unroll
            for (int i = 0; i < 8; ++i) {
                int m = m0 + rr * 16 + mr + i;
                float val = acc[rr][t][i] + bn;
                if (mode <= 1) val = (val > 0.f) ? (val + 1.f) : __expf(val); // elu+1
                int b = m >> 12, s = m & (S_ - 1);
                int h = n >> 6,  d = n & 63;
                if (mode == 0) {
                    ((u16*)out)[((size_t)(b * H_ + h) * S_ + s) * HD_ + d] = f2bf(val);
                } else if (mode <= 2) {
                    ((u16*)out)[((size_t)(b * H_ + h) * HD_ + d) * S_ + s] = f2bf(val);
                } else {
                    ((float*)out)[(size_t)m * D_ + n] = val;
                }
            }
        }
    }
}

// ---- kv[d,e] += sum_s k[s,d] v[s,e] (per b,h, split-S) + k_sum[d] -------
// k_t, v_t stored [B,H,HD,S]; partials accumulated with f32 atomics.
__global__ __launch_bounds__(128) void kv_kernel(
    const u16* __restrict__ kt, const u16* __restrict__ vt,
    float* __restrict__ kv, float* __restrict__ ksum)
{
    const int bh    = blockIdx.x;                  // 0..63
    const int chunk = blockIdx.y;                  // 0..KV_SPLIT-1
    const int wave  = threadIdx.x >> 5;            // 0..3 -> d block
    const int lane  = threadIdx.x & 31;
    const int sbeg  = chunk * (S_ / KV_SPLIT);
    const u16* kb = kt + (size_t)bh * HD_ * S_ + (size_t)(wave * 16) * S_;
    const u16* vb = vt + (size_t)bh * HD_ * S_;

    v8f acc[4] = {};
    float srow = 0.f;                              // row-sum of k -> k_sum

    // pipeline A
    v16bf a = load_a(kb + sbeg, S_);
    for (int s0 = sbeg; s0 < sbeg + S_ / KV_SPLIT; s0 += 32) {
        v16bf an = a;
        if (s0 + 32 < sbeg + S_ / KV_SPLIT) an = load_a(kb + s0 + 32, S_);
        #pragma unroll
        for (int j = 0; j < 16; ++j) srow += (float)a[j];
        #pragma unroll
        for (int t = 0; t < 4; ++t)
            acc[t] = wmma_bf16(a, load_b(vb + (size_t)(t * 16) * S_ + s0, S_), acc[t]);
        a = an;
    }
    // lanes L and L^16 hold the two K-halves of the same row d
    srow += __shfl_xor(srow, 16, 32);
    if (lane < 16) atomicAdd(&ksum[bh * HD_ + wave * 16 + lane], srow);

    const int nl = lane & 15, mr = (lane >> 4) * 8;
    #pragma unroll
    for (int t = 0; t < 4; ++t)
        #pragma unroll
        for (int i = 0; i < 8; ++i)
            atomicAdd(&kv[(size_t)bh * HD_ * HD_ + (wave * 16 + mr + i) * HD_ + t * 16 + nl],
                      acc[t][i]);
}

// ---- kv[d,e] f32 -> kv_t[e,d] bf16 --------------------------------------
__global__ void kvt_kernel(const float* __restrict__ kv, u16* __restrict__ kvt) {
    int bh = blockIdx.x;
    const float* src = kv + (size_t)bh * HD_ * HD_;
    u16* dst = kvt + (size_t)bh * HD_ * HD_;
    for (int i = threadIdx.x; i < HD_ * HD_; i += blockDim.x) {
        int d = i >> 6, e = i & 63;
        dst[e * HD_ + d] = f2bf(src[i]);
    }
}

// ---- z[b,h,s] = 1 / (q[s,:] . k_sum + 1e-6) -----------------------------
__global__ void z_kernel(const u16* __restrict__ q, const float* __restrict__ ksum,
                         float* __restrict__ z)
{
    int idx = blockIdx.x * blockDim.x + threadIdx.x;  // B*H*S
    int bh = idx >> 12, s = idx & (S_ - 1);
    const u16* qr = q + ((size_t)bh * S_ + s) * HD_;
    const float* ks = ksum + bh * HD_;
    float dot = 0.f;
    #pragma unroll
    for (int d = 0; d < HD_; ++d) dot += bf2f(qr[d]) * ks[d];
    z[idx] = 1.f / (dot + 1e-6f);
}

// ---- out[s,e] = z[s] * sum_d q[s,d] kv[d,e]; merge heads -> [B,S,D] bf16
__global__ __launch_bounds__(256) void attn_kernel(
    const u16* __restrict__ q, const u16* __restrict__ kvt,
    const float* __restrict__ z, u16* __restrict__ attn)
{
    const int bh   = blockIdx.y;
    const int wave = threadIdx.x >> 5;
    const int lane = threadIdx.x & 31;
    const int m0 = blockIdx.x * 128 + wave * 16;          // s
    const u16* qb = q + ((size_t)bh * S_ + m0) * HD_;
    const u16* kb = kvt + (size_t)bh * HD_ * HD_;

    v8f acc[4] = {};
    #pragma unroll
    for (int d0 = 0; d0 < HD_; d0 += 32) {
        v16bf a = load_a(qb + d0, HD_);
        #pragma unroll
        for (int t = 0; t < 4; ++t)
            acc[t] = wmma_bf16(a, load_b(kb + t * 16 * HD_ + d0, HD_), acc[t]);
    }

    const int nl = lane & 15, mr = (lane >> 4) * 8;
    const int b = bh >> 4, h = bh & 15;
    #pragma unroll
    for (int i = 0; i < 8; ++i) {
        int s = m0 + mr + i;
        float zz = z[(size_t)bh * S_ + s];
        #pragma unroll
        for (int t = 0; t < 4; ++t) {
            int e = t * 16 + nl;
            attn[(size_t)(b * S_ + s) * D_ + h * HD_ + e] = f2bf(acc[t][i] * zz);
        }
    }
}

// ---- host orchestration --------------------------------------------------
extern "C" void kernel_launch(void* const* d_in, const int* in_sizes, int n_in,
                              void* d_out, int out_size, void* d_ws, size_t ws_size,
                              hipStream_t stream) {
    (void)in_sizes; (void)n_in; (void)out_size; (void)ws_size;
    const float* x  = (const float*)d_in[0];
    const float* Wq = (const float*)d_in[1];
    const float* bq = (const float*)d_in[2];
    const float* Wk = (const float*)d_in[3];
    const float* bk = (const float*)d_in[4];
    const float* Wv = (const float*)d_in[5];
    const float* bv = (const float*)d_in[6];
    const float* Wo = (const float*)d_in[7];
    const float* bo = (const float*)d_in[8];
    float* out = (float*)d_out;

    char* ws = (char*)d_ws;
    size_t off = 0;
    auto alloc = [&](size_t bytes) -> void* {
        void* p = ws + off;
        off += (bytes + 255) & ~(size_t)255;
        return p;
    };
    u16*   xb    = (u16*)  alloc((size_t)MTOT * D_ * 2);   // x bf16 (reused as attn-out)
    u16*   wqb   = (u16*)  alloc((size_t)D_ * D_ * 2);
    u16*   wkb   = (u16*)  alloc((size_t)D_ * D_ * 2);
    u16*   wvb   = (u16*)  alloc((size_t)D_ * D_ * 2);
    u16*   wob   = (u16*)  alloc((size_t)D_ * D_ * 2);
    u16*   qb    = (u16*)  alloc((size_t)MTOT * D_ * 2);   // [B,H,S,HD]
    u16*   ktb   = (u16*)  alloc((size_t)MTOT * D_ * 2);   // [B,H,HD,S]
    u16*   vtb   = (u16*)  alloc((size_t)MTOT * D_ * 2);   // [B,H,HD,S]
    float* kvb   = (float*)alloc((size_t)B_ * H_ * HD_ * HD_ * 4);
    u16*   kvtb  = (u16*)  alloc((size_t)B_ * H_ * HD_ * HD_ * 2);
    float* ksumb = (float*)alloc((size_t)B_ * H_ * HD_ * 4);
    float* zb    = (float*)alloc((size_t)B_ * H_ * S_ * 4);
    u16*   attnb = xb;   // alias: x bf16 no longer needed after V projection

    // 0) zero the atomic accumulators (ws is not re-poisoned between replays)
    zero_kernel<<<(B_ * H_ * HD_ * HD_ + 255) / 256, 256, 0, stream>>>(kvb, B_ * H_ * HD_ * HD_);
    zero_kernel<<<(B_ * H_ * HD_ + 255) / 256, 256, 0, stream>>>(ksumb, B_ * H_ * HD_);

    // 1) fp32 -> bf16 (vectorized streams)
    cvt_kernel<<<2048, 256, 0, stream>>>(x,  xb,  MTOT * D_);
    cvt_kernel<<<512,  256, 0, stream>>>(Wq, wqb, D_ * D_);
    cvt_kernel<<<512,  256, 0, stream>>>(Wk, wkb, D_ * D_);
    cvt_kernel<<<512,  256, 0, stream>>>(Wv, wvb, D_ * D_);
    cvt_kernel<<<512,  256, 0, stream>>>(Wo, wob, D_ * D_);

    // 2) Q/K/V projections (WMMA bf16, async-LDS staged B, fused epilogues)
    dim3 pg(D_ / 64, MTOT / 256);   // (16, 64)
    proj_kernel<<<pg, 256, 0, stream>>>(xb, wqb, bq, qb,  0);
    proj_kernel<<<pg, 256, 0, stream>>>(xb, wkb, bk, ktb, 1);
    proj_kernel<<<pg, 256, 0, stream>>>(xb, wvb, bv, vtb, 2);

    // 3) kv = k^T v and k_sum (per head, split-S, atomic accumulate)
    kv_kernel<<<dim3(B_ * H_, KV_SPLIT), 128, 0, stream>>>(ktb, vtb, kvb, ksumb);
    kvt_kernel<<<B_ * H_, 256, 0, stream>>>(kvb, kvtb);

    // 4) normalizer z = 1/(q . k_sum + 1e-6)
    z_kernel<<<(B_ * H_ * S_) / 256, 256, 0, stream>>>(qb, ksumb, zb);

    // 5) out = z * (q @ kv), merge heads -> [B,S,D] bf16 (into xb alias)
    attn_kernel<<<dim3(S_ / 128, B_ * H_), 256, 0, stream>>>(qb, kvtb, zb, attnb);

    // 6) final projection -> f32 output
    proj_kernel<<<pg, 256, 0, stream>>>(attnb, wob, bo, out, 3);
}
